// RiemannianParallelAttention_55671366091637
// MI455X (gfx1250) — matrix operand order repaired and checked by reading, not compiled, not run on hardware
//
#include <hip/hip_runtime.h>
#include <hip/hip_bf16.h>

#define USE_ASYNC_LDS 1

typedef __bf16 bf16;
typedef __attribute__((ext_vector_type(16))) __bf16 v16bf;
typedef __attribute__((ext_vector_type(8)))  __bf16 v8bf;
typedef __attribute__((ext_vector_type(8)))  float  v8f;

#define WMMA_BF16(a, b, c) \
  __builtin_amdgcn_wmma_f32_16x16x32_bf16(false, (a), false, (b), (short)0, (c), false, false)

// ---- helpers -------------------------------------------------------

// Load 16 contiguous bf16 (32 bytes, 16B-aligned) as one WMMA fragment.
__device__ inline v16bf frag_load32(const bf16* p) {
  const v8bf lo = *(const v8bf*)p;
  const v8bf hi = *(const v8bf*)(p + 8);
  return __builtin_shufflevector(lo, hi, 0, 1, 2, 3, 4, 5, 6, 7, 8, 9, 10, 11,
                                 12, 13, 14, 15);
}

// A 16x32 fragment swizzle: element (m,k) -> lane/pos (contiguous k order):
// lanes 0-15 hold K 0..15, lanes 16-31 hold K 16..31.
__device__ inline int a_swz_lane(int m, int k) { return m + ((k >> 4) << 4); }
__device__ inline int a_swz_pos(int k)         { return k & 15; }

// Pack two floats to bf16x2 and store as one dword (dst must be 4B-aligned).
__device__ inline void store_pk(bf16* dst, float a, float b) {
  union { bf16 h[2]; unsigned u; } pk;
  pk.h[0] = (bf16)a;
  pk.h[1] = (bf16)b;
  *(unsigned*)dst = pk.u;
}

// =====================================================================
// GEMM: Out(M,N) = A(M,K) @ W(N,K)^T + bias(N).  64x64 tile per block,
// 128 threads = 4 waves.  Register-staged pipeline: next k-tile's global
// loads are issued while the current tile's WMMAs run from LDS.
// =====================================================================
template <bool OUT_BF16>
__global__ __launch_bounds__(128) void gemm_bias_kernel(
    const float* __restrict__ A, const float* __restrict__ W,
    const float* __restrict__ bias, void* __restrict__ outp,
    int Mdim, int Ndim, int Kdim) {
  __shared__ __align__(16) bf16 sA[4][32][16];  // swizzled A fragments per 16-row block
  __shared__ __align__(16) bf16 sB[64][40];     // rows = n, cols = k (80B stride)
  const int tid = threadIdx.x;
  const int lane = tid & 31;
  const int wave = tid >> 5;
  const int n0 = blockIdx.x * 64;
  const int m0 = blockIdx.y * 64;

  const v8f vz = {0.f, 0.f, 0.f, 0.f, 0.f, 0.f, 0.f, 0.f};
  v8f acc[4];
#pragma unroll
  for (int t = 0; t < 4; ++t) acc[t] = vz;

  float2 ra[8], rw[8];
  auto load_stage = [&](int k0) {
#pragma unroll
    for (int it = 0; it < 8; ++it) {
      const int e = tid + it * 128;
      const int i = e >> 4;              // row 0..63
      const int kp = (e & 15) << 1;      // even k 0..30
      ra[it] = *(const float2*)&A[(size_t)(m0 + i) * Kdim + k0 + kp];
      rw[it] = *(const float2*)&W[(size_t)(n0 + i) * Kdim + k0 + kp];
    }
  };

  load_stage(0);
  for (int k0 = 0; k0 < Kdim; k0 += 32) {
#pragma unroll
    for (int it = 0; it < 8; ++it) {     // staged regs -> LDS (pk-bf16 dwords)
      const int e = tid + it * 128;
      const int i = e >> 4;
      const int kp = (e & 15) << 1;
      store_pk(&sA[i >> 4][a_swz_lane(i & 15, kp)][a_swz_pos(kp)], ra[it].x, ra[it].y);
      store_pk(&sB[i][kp], rw[it].x, rw[it].y);
    }
    __syncthreads();
    if (k0 + 32 < Kdim) load_stage(k0 + 32);  // overlap with WMMAs below

    const v16bf af = frag_load32(&sA[wave][lane][0]);
    const int kb = (lane & 16) ? 16 : 0;
    const int nr = lane & 15;
#pragma unroll
    for (int t = 0; t < 4; ++t) {
      const v16bf bf = frag_load32(&sB[t * 16 + nr][kb]);
      acc[t] = WMMA_BF16(af, bf, acc[t]);
    }
    __syncthreads();
  }

  const int nlo = lane & 15;
  const int mhi = (lane & 16) ? 8 : 0;
#pragma unroll
  for (int t = 0; t < 4; ++t) {
    const int n = n0 + t * 16 + nlo;
    const float bv = bias[n];
#pragma unroll
    for (int r = 0; r < 8; ++r) {
      const int m = m0 + wave * 16 + r + mhi;
      const float v = acc[t][r] + bv;
      const size_t idx = (size_t)m * Ndim + n;
      if (OUT_BF16) ((bf16*)outp)[idx] = (bf16)v;
      else          ((float*)outp)[idx] = v;
    }
  }
}

// =====================================================================
// Gram: S[bn] = xs @ xs^T / 512 + eps*I, xs = 64x512 contiguous rows.
// Same pipeline; one tile staged twice (swizzled A + row-major B^T).
// =====================================================================
__global__ __launch_bounds__(128) void gram_kernel(const float* __restrict__ x,
                                                   float* __restrict__ S) {
  __shared__ __align__(16) bf16 sXa[4][32][16];
  __shared__ __align__(16) bf16 sXb[64][40];
  const int tid = threadIdx.x, lane = tid & 31, wave = tid >> 5;
  const size_t bn = blockIdx.x;
  const float* xs = x + bn * 64 * 512;

  const v8f vz = {0.f, 0.f, 0.f, 0.f, 0.f, 0.f, 0.f, 0.f};
  v8f acc[4];
#pragma unroll
  for (int t = 0; t < 4; ++t) acc[t] = vz;

  float2 rx[8];
  auto load_stage = [&](int k0) {
#pragma unroll
    for (int it = 0; it < 8; ++it) {
      const int e = tid + it * 128;
      const int i = e >> 4;
      const int kp = (e & 15) << 1;
      rx[it] = *(const float2*)&xs[(size_t)i * 512 + k0 + kp];
    }
  };

  load_stage(0);
  for (int k0 = 0; k0 < 512; k0 += 32) {
#pragma unroll
    for (int it = 0; it < 8; ++it) {
      const int e = tid + it * 128;
      const int i = e >> 4;
      const int kp = (e & 15) << 1;
      store_pk(&sXa[i >> 4][a_swz_lane(i & 15, kp)][a_swz_pos(kp)], rx[it].x, rx[it].y);
      store_pk(&sXb[i][kp], rx[it].x, rx[it].y);
    }
    __syncthreads();
    if (k0 + 32 < 512) load_stage(k0 + 32);

    const v16bf af = frag_load32(&sXa[wave][lane][0]);
    const int kb = (lane & 16) ? 16 : 0;
    const int nr = lane & 15;
#pragma unroll
    for (int t = 0; t < 4; ++t) {
      const v16bf bf = frag_load32(&sXb[t * 16 + nr][kb]);
      acc[t] = WMMA_BF16(af, bf, acc[t]);
    }
    __syncthreads();
  }

  float* Sout = S + bn * 4096;
  const int nlo = lane & 15;
  const int mhi = (lane & 16) ? 8 : 0;
#pragma unroll
  for (int t = 0; t < 4; ++t) {
#pragma unroll
    for (int r = 0; r < 8; ++r) {
      const int mi = wave * 16 + r + mhi;
      const int j  = t * 16 + nlo;
      Sout[mi * 64 + j] = acc[t][r] * (1.f / 512.f) + ((mi == j) ? 1e-5f : 0.f);
    }
  }
}

// =====================================================================
// SPD log via parallel cyclic Jacobi, in-place S -> logm(S).
// =====================================================================
__global__ __launch_bounds__(64) void spd_log_kernel(float* __restrict__ S) {
  __shared__ float Am[64][65];
  __shared__ float Qm[64][65];
  __shared__ float cs[32], sn[32];
  __shared__ int   pp[32], qq[32];
  __shared__ float lw[64];
  const int t = threadIdx.x;
  float* Sb = S + (size_t)blockIdx.x * 4096;

  for (int j = 0; j < 64; ++j) {
    Am[t][j] = Sb[t * 64 + j];
    Qm[t][j] = (t == j) ? 1.f : 0.f;
  }
  __syncthreads();

  for (int sweep = 0; sweep < 8; ++sweep) {
    for (int r = 0; r < 63; ++r) {
      if (t < 32) {
        int p, q;
        if (t == 0) { p = 63; q = r % 63; }
        else        { p = (r + t) % 63; q = (r + 63 - t) % 63; }
        if (p < q) { const int tmp = p; p = q; q = tmp; }
        const float apq = Am[p][q];
        float c = 1.f, s = 0.f;
        if (fabsf(apq) > 1e-12f) {
          const float th = (Am[q][q] - Am[p][p]) / (2.f * apq);
          const float tt = copysignf(1.f, th) / (fabsf(th) + sqrtf(1.f + th * th));
          c = rsqrtf(1.f + tt * tt);
          s = tt * c;
        }
        cs[t] = c; sn[t] = s; pp[t] = p; qq[t] = q;
      }
      __syncthreads();
#pragma unroll 4
      for (int j = 0; j < 32; ++j) {  // A <- A*J, Q <- Q*J (thread t = row t)
        const int p = pp[j], q = qq[j];
        const float c = cs[j], s = sn[j];
        const float ap = Am[t][p], aq = Am[t][q];
        Am[t][p] = c * ap - s * aq;
        Am[t][q] = s * ap + c * aq;
        const float up = Qm[t][p], uq = Qm[t][q];
        Qm[t][p] = c * up - s * uq;
        Qm[t][q] = s * up + c * uq;
      }
      __syncthreads();
#pragma unroll 4
      for (int j = 0; j < 32; ++j) {  // A <- J^T*A (thread t = column t)
        const int p = pp[j], q = qq[j];
        const float c = cs[j], s = sn[j];
        const float ap = Am[p][t], aq = Am[q][t];
        Am[p][t] = c * ap - s * aq;
        Am[q][t] = s * ap + c * aq;
      }
      __syncthreads();
    }
  }

  lw[t] = logf(fmaxf(Am[t][t], 1e-5f));
  __syncthreads();
  for (int e = 0; e < 64; ++e) Am[t][e] = Qm[t][e] * lw[e];  // R = Q diag(log w)
  __syncthreads();
  for (int j = 0; j < 64; ++j) {                             // Lm = R Q^T
    float acc = 0.f;
    for (int e = 0; e < 64; ++e) acc += Am[t][e] * Qm[j][e];
    Sb[t * 64 + j] = acc;
  }
}

// =====================================================================
// Attention block (64x64x64): one block per (b, c|n, head-in-branch).
// 256 threads = 8 waves; Q/P staged fragment-major, K row-major
// (filled via async global->LDS b128 DMA), V transposed at fill.
// =====================================================================
__global__ __launch_bounds__(256) void attn_kernel(
    const bf16* __restrict__ qkv, const float* __restrict__ Lm,
    const float* __restrict__ head_scales, float* __restrict__ attn_out,
    int spatial) {
  __shared__ __align__(16) bf16 sQa[4][2][32][16];  // swizzled Q fragments
  __shared__ __align__(16) bf16 sK[64][40];         // rows = key, cols = d
  __shared__ __align__(16) bf16 sVt[64][40];        // rows = d, cols = key
  __shared__ __align__(16) bf16 sPa[4][2][32][16];  // swizzled P fragments
  __shared__ float sS[64][65];

  const int tid = threadIdx.x, lane = tid & 31, wave = tid >> 5;
  const int h = blockIdx.x & 3;
  const int outer = blockIdx.x >> 2;     // b*64 + (c|n)
  const int b  = outer >> 6;
  const int cn = outer & 63;
  const int hg = spatial ? (4 + h) : h;  // global head index
  const float hs = spatial ? head_scales[h] : 0.f;
  const float* LmB = Lm + (size_t)outer * 4096;

  // ---- stage K: async global->LDS, 16B per lane per issue ----
#if USE_ASYNC_LDS
  {
    const unsigned lds_base = (unsigned)(size_t)&sK[0][0];
#pragma unroll
    for (int it = 0; it < 2; ++it) {
      const int c = tid + it * 256;
      const int i = c >> 3, g = c & 7;  // row, 8-dim group
      const int l = spatial ? (cn * 64 + i) : (i * 64 + cn);
      const bf16* gsrc =
          qkv + ((size_t)b * 4096 + l) * 1536 + (size_t)hg * 64 + 512 + g * 8;
      const unsigned ldst = lds_base + (unsigned)(i * 80 + g * 16);
      asm volatile("global_load_async_to_lds_b128 %0, %1, off"
                   :: "v"(ldst), "v"((unsigned long long)(size_t)gsrc)
                   : "memory");
    }
  }
#endif

  // ---- stage Q (swizzled) and V (transposed): batched reg staging ----
  unsigned rq[8], rv[8];
#pragma unroll
  for (int it = 0; it < 8; ++it) {
    const int e = tid + it * 256;
    const int i = e >> 5;
    const int kp = (e & 31) << 1;       // two consecutive dims
    const int l = spatial ? (cn * 64 + i) : (i * 64 + cn);
    const size_t rb = ((size_t)b * 4096 + l) * 1536 + (size_t)hg * 64;
    rq[it] = *(const unsigned*)(qkv + rb + kp);
    rv[it] = *(const unsigned*)(qkv + rb + 1024 + kp);
  }
  if (spatial)  // bias tile is read right after the score WMMAs
    __builtin_prefetch(LmB + tid * 16, 0, 1);
#pragma unroll
  for (int it = 0; it < 8; ++it) {
    const int e = tid + it * 256;
    const int i = e >> 5;
    const int kp = (e & 31) << 1;
    *(unsigned*)&sQa[i >> 4][kp >> 5][a_swz_lane(i & 15, kp & 31)][a_swz_pos(kp & 31)] = rq[it];
    union { unsigned u; bf16 hh[2]; } v2;
    v2.u = rv[it];
    sVt[kp][i]     = v2.hh[0];
    sVt[kp + 1][i] = v2.hh[1];
  }

#if USE_ASYNC_LDS
  asm volatile("s_wait_asynccnt 0x0" ::: "memory");
#else
#pragma unroll
  for (int it = 0; it < 8; ++it) {
    const int e = tid + it * 256;
    const int i = e >> 5;
    const int kp = (e & 31) << 1;
    const int l = spatial ? (cn * 64 + i) : (i * 64 + cn);
    const bf16* src =
        qkv + ((size_t)b * 4096 + l) * 1536 + (size_t)hg * 64 + 512 + kp;
    *(unsigned*)&sK[i][kp] = *(const unsigned*)src;
  }
#endif
  __syncthreads();

  const int wm = wave & 3;   // row block
  const int wn = wave >> 2;  // col half
  const int kb = (lane & 16) ? 16 : 0;
  const int nr = lane & 15;
  const v8f vz = {0.f, 0.f, 0.f, 0.f, 0.f, 0.f, 0.f, 0.f};
  v8f a0 = vz, a1 = vz;

  // ---- scores = Q @ K^T ----
#pragma unroll
  for (int kh = 0; kh < 2; ++kh) {
    const v16bf af = frag_load32(&sQa[wm][kh][lane][0]);
    const v16bf b0 = frag_load32(&sK[wn * 32 + nr][kh * 32 + kb]);
    const v16bf b1 = frag_load32(&sK[wn * 32 + 16 + nr][kh * 32 + kb]);
    a0 = WMMA_BF16(af, b0, a0);
    a1 = WMMA_BF16(af, b1, a1);
  }

  const float scale = 0.125f;  // 1/sqrt(64)
  const int mhi = (lane & 16) ? 8 : 0;
#pragma unroll
  for (int r = 0; r < 8; ++r) {
    const int mi = wm * 16 + r + mhi;
    const int c0 = wn * 32 + nr;
    const int c1 = c0 + 16;
    float v0 = a0[r] * scale;
    float v1 = a1[r] * scale;
    if (spatial) {
      v0 += hs * LmB[mi * 64 + c0];
      v1 += hs * LmB[mi * 64 + c1];
    }
    sS[mi][c0] = v0;
    sS[mi][c1] = v1;
  }
  __syncthreads();

  // ---- softmax per row; write P straight into fragment layout ----
  if (tid < 64) {
    float mx = -3.0e38f;
    for (int j = 0; j < 64; ++j) mx = fmaxf(mx, sS[tid][j]);
    float sum = 0.f;
    for (int j = 0; j < 64; ++j) {
      const float e = __expf(sS[tid][j] - mx);
      sS[tid][j] = e;
      sum += e;
    }
    const float inv = 1.f / sum;
    const int m = tid & 15, blk = tid >> 4;
    for (int kp = 0; kp < 64; kp += 2) {
      const int kh = kp >> 5, kk = kp & 31;
      store_pk(&sPa[blk][kh][a_swz_lane(m, kk)][a_swz_pos(kk)],
               sS[tid][kp] * inv, sS[tid][kp + 1] * inv);
    }
  }
  __syncthreads();

  // ---- out = P @ V ----
  a0 = vz; a1 = vz;
#pragma unroll
  for (int kh = 0; kh < 2; ++kh) {
    const v16bf af = frag_load32(&sPa[wm][kh][lane][0]);
    const v16bf b0 = frag_load32(&sVt[wn * 32 + nr][kh * 32 + kb]);
    const v16bf b1 = frag_load32(&sVt[wn * 32 + 16 + nr][kh * 32 + kb]);
    a0 = WMMA_BF16(af, b0, a0);
    a1 = WMMA_BF16(af, b1, a1);
  }

#pragma unroll
  for (int r = 0; r < 8; ++r) {
    const int mi = wm * 16 + r + mhi;
    const int l = spatial ? (cn * 64 + mi) : (mi * 64 + cn);
    const size_t base =
        ((size_t)b * 4096 + l) * 512 + (size_t)hg * 64 + wn * 32 + nr;
    attn_out[base]      = a0[r];
    attn_out[base + 16] = a1[r];
  }
}

// =====================================================================
extern "C" void kernel_launch(void* const* d_in, const int* in_sizes, int n_in,
                              void* d_out, int out_size, void* d_ws, size_t ws_size,
                              hipStream_t stream) {
  (void)in_sizes; (void)n_in; (void)out_size; (void)ws_size;
  const float* x     = (const float*)d_in[0];
  const float* qkvw  = (const float*)d_in[1];
  const float* qkvb  = (const float*)d_in[2];
  const float* fcw   = (const float*)d_in[3];
  const float* fcb   = (const float*)d_in[4];
  const float* hscal = (const float*)d_in[5];

  const size_t M = 65536;  // B*L
  // workspace: qkv bf16 (201 MB) | attn f32 (134 MB) | S/Lm f32 (16.8 MB)
  bf16*  qkv  = (bf16*)d_ws;
  float* attn = (float*)((char*)d_ws + M * 1536 * sizeof(bf16));
  float* Sm   = attn + M * 512;

  gemm_bias_kernel<true><<<dim3(1536 / 64, M / 64), 128, 0, stream>>>(
      x, qkvw, qkvb, (void*)qkv, (int)M, 1536, 512);
  gram_kernel<<<1024, 128, 0, stream>>>(x, Sm);
  spd_log_kernel<<<1024, 64, 0, stream>>>(Sm);
  attn_kernel<<<4096, 256, 0, stream>>>(qkv, Sm, hscal, attn, 0);
  attn_kernel<<<4096, 256, 0, stream>>>(qkv, Sm, hscal, attn, 1);
  gemm_bias_kernel<false><<<dim3(512 / 64, M / 64), 128, 0, stream>>>(
      attn, fcw, fcb, d_out, (int)M, 512, 512);
}
